// CollaborativeExpertsModule_27041114095959
// MI455X (gfx1250) — matrix operand order repaired and checked by reading, not compiled, not run on hardware
//
#include <hip/hip_runtime.h>
#include <math.h>

// ---------------- problem constants ----------------
#define BB 4
#define SS 2048
#define DD 768
#define EE 8
#define KKK 2
#define HH 2048
#define NHEAD 12
#define HDIM 64
#define BS (BB*SS)        // 8192 tokens
#define TD3 (3*DD)        // 2304

// ---------------- types ----------------
typedef __bf16 bf16_t;
typedef bf16_t v8bf  __attribute__((ext_vector_type(8)));
typedef bf16_t v16bf __attribute__((ext_vector_type(16)));
typedef float  v8f   __attribute__((ext_vector_type(8)));

static_assert(sizeof(bf16_t) == 2, "bf16 size");

static __device__ __forceinline__ bf16_t f2bf(float f) {
    unsigned u = __float_as_uint(f);
    unsigned r = u + 0x7FFFu + ((u >> 16) & 1u);   // round to nearest even
    unsigned short h = (unsigned short)(r >> 16);
    bf16_t b;
    __builtin_memcpy(&b, &h, 2);
    return b;
}
static __device__ __forceinline__ float bf2f(bf16_t b) {
    unsigned short h;
    __builtin_memcpy(&h, &b, 2);
    return __uint_as_float(((unsigned)h) << 16);
}
static __device__ __forceinline__ v8f zero8() {
    v8f z;
    for (int i = 0; i < 8; ++i) z[i] = 0.f;
    return z;
}
static __device__ __forceinline__ v8f wmma_bf16(v16bf a, v16bf b, v8f c) {
    return __builtin_amdgcn_wmma_f32_16x16x32_bf16(false, a, false, b, (short)0, c,
                                                   false, false);
}
static __device__ __forceinline__ v16bf load_a16(const bf16_t* arow, int k) {
    v8bf a0 = *(const v8bf*)(arow + k);
    v8bf a1 = *(const v8bf*)(arow + k + 8);
    return __builtin_shufflevector(a0, a1, 0, 1, 2, 3, 4, 5, 6, 7,
                                   8, 9, 10, 11, 12, 13, 14, 15);
}

// One 16x16 fp32 C tile per wave. A: 16 rows in LDS (row-major, lda elems).
// B: global bf16, row-major [N, ldb], row n = n0 + (lane&15).
// A lane layout: lanes 0-15 rows 0-15 with K 0..15 contiguous; lanes 16-31 K 16..31.
// B lane layout: lanes 0-15 K 0..15 contiguous; lanes 16-31 K 16..31.
static __device__ __forceinline__ v8f gemm_tile(const bf16_t* As, int lda,
                                                const bf16_t* Bg, int ldb,
                                                int n0, int klen, int koffB,
                                                int lane, v8f acc) {
    const int m    = lane & 15;
    const int aoff = (lane & 16) ? 16 : 0;
    const int boff = (lane & 16) ? 16 : 0;
    const bf16_t* arow = As + m * lda + aoff;
    const bf16_t* brow = Bg + (size_t)(n0 + m) * ldb + koffB + boff;
    for (int k = 0; k < klen; k += 32) {
        v16bf va = load_a16(arow, k);
        v16bf vb = *(const v16bf*)(brow + k);
        acc = wmma_bf16(va, vb, acc);
    }
    return acc;
}

// 2 M-subtiles sharing one B stream (halves L2 weight traffic)
static __device__ __forceinline__ void gemm_tile_m2(const bf16_t* As0, const bf16_t* As1,
                                                    int lda, const bf16_t* Bg, int ldb,
                                                    int n0, int klen, int koffB, int lane,
                                                    v8f* c0, v8f* c1) {
    const int m    = lane & 15;
    const int aoff = (lane & 16) ? 16 : 0;
    const int boff = (lane & 16) ? 16 : 0;
    const bf16_t* a0r = As0 + m * lda + aoff;
    const bf16_t* a1r = As1 + m * lda + aoff;
    const bf16_t* brow = Bg + (size_t)(n0 + m) * ldb + koffB + boff;
    for (int k = 0; k < klen; k += 32) {
        v16bf vb = *(const v16bf*)(brow + k);
        *c0 = wmma_bf16(load_a16(a0r, k), vb, *c0);
        *c1 = wmma_bf16(load_a16(a1r, k), vb, *c1);
    }
}

// 2 M-subtiles x 2 B streams (w1,w3): 4 WMMA per K-step, every load reused twice
static __device__ __forceinline__ void gemm_tile2_m2(const bf16_t* As0, const bf16_t* As1,
                                                     int lda, const bf16_t* B1,
                                                     const bf16_t* B3, int ldb, int n0,
                                                     int klen, int lane,
                                                     v8f* c10, v8f* c11,
                                                     v8f* c30, v8f* c31) {
    const int m    = lane & 15;
    const int aoff = (lane & 16) ? 16 : 0;
    const int boff = (lane & 16) ? 16 : 0;
    const bf16_t* a0r = As0 + m * lda + aoff;
    const bf16_t* a1r = As1 + m * lda + aoff;
    const bf16_t* b1r = B1 + (size_t)(n0 + m) * ldb + boff;
    const bf16_t* b3r = B3 + (size_t)(n0 + m) * ldb + boff;
    for (int k = 0; k < klen; k += 32) {
        v16bf va0 = load_a16(a0r, k);
        v16bf va1 = load_a16(a1r, k);
        v16bf vb1 = *(const v16bf*)(b1r + k);
        v16bf vb3 = *(const v16bf*)(b3r + k);
        *c10 = wmma_bf16(va0, vb1, *c10);
        *c11 = wmma_bf16(va1, vb1, *c11);
        *c30 = wmma_bf16(va0, vb3, *c30);
        *c31 = wmma_bf16(va1, vb3, *c31);
    }
}

// ---------------- cast / init kernels ----------------
__global__ void cast_f32_bf16(const float* __restrict__ src, bf16_t* __restrict__ dst,
                              long n) {
    long i = (long)blockIdx.x * blockDim.x + threadIdx.x;
    long stride = (long)gridDim.x * blockDim.x;
    for (; i < n; i += stride) dst[i] = f2bf(src[i]);
}

__global__ void zero_stats_kernel(float* s) {
    if (threadIdx.x < 16) s[threadIdx.x] = 0.f;
}

// ---------------- FFN: all-expert fused SwiGLU ----------------
// grid: (BS/32, E), block 256 (8 waves). M-tile = 32 tokens (2 sub-tiles of 16)
// so every weight load from L2 feeds two WMMAs. eo[token][e][d] fp32.
__global__ __launch_bounds__(256) void ffn_kernel(const bf16_t* __restrict__ xbf,
                                                  const bf16_t* __restrict__ w1bf,
                                                  const bf16_t* __restrict__ w3bf,
                                                  const bf16_t* __restrict__ w2bf,
                                                  float* __restrict__ eo) {
    __shared__ __align__(32) bf16_t sX[32 * DD];    // 48 KB
    __shared__ __align__(32) bf16_t sH[32 * 256];   // 16 KB (HID chunk = 256)
    const int tid = threadIdx.x, lane = tid & 31, wave = tid >> 5;
    const int mbase = blockIdx.x * 32;
    const int e = blockIdx.y;

    {   // stage 32 contiguous token rows of x (bf16) into LDS
        const uint4* src = (const uint4*)(xbf + (size_t)mbase * DD);
        uint4* dst = (uint4*)sX;
        for (int i = tid; i < 32 * DD * 2 / 16; i += 256) dst[i] = src[i];
    }
    __syncthreads();

    const bf16_t* w1e = w1bf + (size_t)e * HH * DD;
    const bf16_t* w3e = w3bf + (size_t)e * HH * DD;
    const bf16_t* w2e = w2bf + (size_t)e * DD * HH;
    const bf16_t* sX1 = sX + 16 * DD;               // second M sub-tile

    v8f c2[12];                                     // [j] rows 0-15, [6+j] rows 16-31
    for (int j = 0; j < 12; ++j) c2[j] = zero8();

    for (int hc = 0; hc < 8; ++hc) {                // HID in chunks of 256
        const int hidb = hc * 256;
        // GEMM1: h = silu(x@w1^T) * (x@w3^T) for this HID chunk
        for (int t = wave; t < 16; t += 8) {
            const int n0 = hidb + t * 16;
            v8f a10 = zero8(), a11 = zero8(), a30 = zero8(), a31 = zero8();
            gemm_tile2_m2(sX, sX1, DD, w1e, w3e, DD, n0, DD, lane,
                          &a10, &a11, &a30, &a31);
            const int colb = t * 16 + (lane & 15);
            const int rowoff = (lane & 16) ? 8 : 0;
            for (int i = 0; i < 8; ++i) {
                float g0 = a10[i];
                float h0 = (g0 / (1.f + __expf(-g0))) * a30[i];
                sH[(i + rowoff) * 256 + colb] = f2bf(h0);
                float g1 = a11[i];
                float h1 = (g1 / (1.f + __expf(-g1))) * a31[i];
                sH[(16 + i + rowoff) * 256 + colb] = f2bf(h1);
            }
        }
        __syncthreads();
        // GEMM2 partial accumulate: eo += h_chunk @ w2_chunk^T (accs in VGPRs)
        int j = 0;
        for (int t = wave; t < 48; t += 8, ++j)
            gemm_tile_m2(sH, sH + 16 * 256, 256, w2e, HH, t * 16, 256, hidb, lane,
                         &c2[j], &c2[6 + j]);
        __syncthreads();
    }
    // epilogue: fp32 expert outputs for both 16-row sub-tiles
    int j = 0;
    for (int t = wave; t < 48; t += 8, ++j) {
        const int n = t * 16 + (lane & 15);
        const int rowoff = (lane & 16) ? 8 : 0;
        for (int i = 0; i < 8; ++i) {
            const int tok0 = mbase + i + rowoff;
            eo[((size_t)tok0 * EE + e) * DD + n] = c2[j][i];
            const int tok1 = mbase + 16 + i + rowoff;
            eo[((size_t)tok1 * EE + e) * DD + n] = c2[6 + j][i];
        }
    }
}

// ---------------- router: fused MHA(E=8) + RMSNorm + gate + top-2 ----------------
// grid: BS/2 (2 tokens -> 16 rows), block 256
__global__ __launch_bounds__(256) void router_kernel(
    const float* __restrict__ eo, const bf16_t* __restrict__ rinw,
    const float* __restrict__ rinb, const bf16_t* __restrict__ routw,
    const float* __restrict__ routb, const float* __restrict__ rnormw,
    const float* __restrict__ gatew, const float* __restrict__ gateb,
    int* __restrict__ topk_idx, float* __restrict__ topk_p,
    float* __restrict__ usage) {
    __shared__ __align__(32) bf16_t sA[16 * DD];     // expert_flat tile; later ctx
    __shared__ __align__(32) bf16_t sO[16 * DD];     // attention output
    __shared__ __align__(32) bf16_t sQKV[16 * 192];  // per-head q|k|v
    __shared__ float sSum[2 * DD];
    __shared__ float sRowSq[16], sScale[16], sLog[16];
    const int tid = threadIdx.x, lane = tid & 31, wave = tid >> 5;
    const int t0 = blockIdx.x * 2;
    const size_t gbase = (size_t)t0 * EE * DD;   // 16 contiguous rows of [BS*E, D]

    for (int i = tid; i < 16 * DD; i += 256) sA[i] = f2bf(eo[gbase + i]);
    if (tid < 16) { sRowSq[tid] = 0.f; }
    __syncthreads();

    // qkv + attention, one head at a time (keeps LDS small)
    for (int h = 0; h < NHEAD; ++h) {
        for (int t = wave; t < 12; t += 8) {
            const int p = t >> 2, sub = t & 3;             // 0=q 1=k 2=v
            const int n0 = p * DD + h * HDIM + sub * 16;
            v8f acc = zero8();
            acc = gemm_tile(sA, DD, rinw, DD, n0, DD, 0, lane, acc);
            const int n = n0 + (lane & 15);
            const float bias = rinb[n];
            const int colb = p * HDIM + sub * 16 + (lane & 15);
            const int rowoff = (lane & 16) ? 8 : 0;
            for (int i = 0; i < 8; ++i)
                sQKV[(i + rowoff) * 192 + colb] = f2bf(acc[i] + bias);
        }
        __syncthreads();
        if (tid < 16) {                                    // 2 tokens x 8 rows
            const int tk = tid >> 3, i = tid & 7;
            const bf16_t* q = sQKV + (tk * 8 + i) * 192;
            float sc[8], mx = -1e30f;
            for (int jj = 0; jj < 8; ++jj) {
                const bf16_t* kv = sQKV + (tk * 8 + jj) * 192 + 64;
                float s = 0.f;
                for (int c = 0; c < 64; ++c) s += bf2f(q[c]) * bf2f(kv[c]);
                s *= 0.125f;                                // 1/sqrt(64)
                sc[jj] = s; mx = fmaxf(mx, s);
            }
            float z = 0.f;
            for (int jj = 0; jj < 8; ++jj) { sc[jj] = __expf(sc[jj] - mx); z += sc[jj]; }
            const float inv = 1.f / z;
            bf16_t* orow = sO + (tk * 8 + i) * DD + h * HDIM;
            for (int c = 0; c < 64; ++c) {
                float o = 0.f;
                for (int jj = 0; jj < 8; ++jj)
                    o += sc[jj] * bf2f(sQKV[(tk * 8 + jj) * 192 + 128 + c]);
                orow[c] = f2bf(o * inv);
            }
        }
        __syncthreads();
    }

    // out projection + residual + row sum-of-squares (ctx overlays sA)
    bf16_t* sCtx = sA;
    for (int t = wave; t < 48; t += 8) {
        const int n0 = t * 16;
        v8f acc = zero8();
        acc = gemm_tile(sO, DD, routw, DD, n0, DD, 0, lane, acc);
        const int n = n0 + (lane & 15);
        const float bias = routb[n];
        const int rowoff = (lane & 16) ? 8 : 0;
        for (int i = 0; i < 8; ++i) {
            const int r = i + rowoff;
            const float v = acc[i] + bias + eo[gbase + (size_t)r * DD + n];
            sCtx[r * DD + n] = f2bf(v);
            atomicAdd(&sRowSq[r], v * v);
        }
    }
    __syncthreads();
    if (tid < 16) sScale[tid] = rsqrtf(sRowSq[tid] / (float)DD + 1e-5f);
    if (tid < 16) sLog[tid] = 0.f;
    __syncthreads();
    // rmsnorm + mean over experts -> token summary
    for (int idx = tid; idx < 2 * DD; idx += 256) {
        const int tk = idx / DD, d = idx - tk * DD;
        float s = 0.f;
        for (int r = 0; r < 8; ++r)
            s += bf2f(sCtx[(tk * 8 + r) * DD + d]) * sScale[tk * 8 + r];
        sSum[idx] = s * (rnormw[d] * 0.125f);              // *w[d], /8 for the mean
    }
    __syncthreads();
    {   // gate logits: 16 (token,expert) dots of 768, 16 threads each
        const int g = tid >> 4, part = tid & 15;
        const int tk = g >> 3, e = g & 7;
        const float* gw = gatew + e * DD;
        const float* su = sSum + tk * DD;
        float s = 0.f;
        for (int d = part * 48; d < part * 48 + 48; ++d) s += su[d] * gw[d];
        atomicAdd(&sLog[g], s);
    }
    __syncthreads();
    if (tid < 2) {
        const int tk = tid;
        float l[8], mx = -1e30f;
        for (int e = 0; e < 8; ++e) { l[e] = sLog[tk * 8 + e] + gateb[e]; mx = fmaxf(mx, l[e]); }
        float z = 0.f, p[8];
        for (int e = 0; e < 8; ++e) { p[e] = __expf(l[e] - mx); z += p[e]; }
        const float inv = 1.f / z;
        for (int e = 0; e < 8; ++e) atomicAdd(&usage[e], p[e] * inv);
        int i1 = 0;
        for (int e = 1; e < 8; ++e) if (l[e] > l[i1]) i1 = e;
        int i2 = (i1 == 0) ? 1 : 0;
        for (int e = 0; e < 8; ++e) if (e != i1 && l[e] > l[i2]) i2 = e;
        const float q1 = 1.f / (1.f + __expf(l[i2] - l[i1]));
        const int tokg = t0 + tk;
        topk_idx[tokg * 2 + 0] = i1; topk_idx[tokg * 2 + 1] = i2;
        topk_p[tokg * 2 + 0] = q1;   topk_p[tokg * 2 + 1] = 1.f - q1;
    }
}

// ---------------- collab: gather + MHA(K=2) + RMSNorm + MLP + combine + out-proj --
// grid: BS/8 (8 tokens -> 16 rows), block 256
__global__ __launch_bounds__(256) void collab_kernel(
    const float* __restrict__ eo, const int* __restrict__ topk_idx,
    const float* __restrict__ topk_p, const bf16_t* __restrict__ cinw,
    const float* __restrict__ cinb, const bf16_t* __restrict__ coutw,
    const float* __restrict__ coutb, const float* __restrict__ cnormw,
    const bf16_t* __restrict__ f1w, const float* __restrict__ f1b,
    const bf16_t* __restrict__ f2w, const float* __restrict__ f2b,
    const bf16_t* __restrict__ oww, float* __restrict__ out,
    float* __restrict__ entacc) {
    __shared__ __align__(32) bf16_t sA[16 * DD];     // sel tile; later collab
    __shared__ __align__(32) bf16_t sO[16 * DD];     // attn out; later t1; later y
    __shared__ __align__(32) bf16_t sQKV[16 * 192];
    __shared__ int   sRowBase[16];
    __shared__ float sP[16];
    __shared__ float sRowSq[16], sScale[16];
    __shared__ float sAttn[32];
    const int tid = threadIdx.x, lane = tid & 31, wave = tid >> 5;
    const int t0 = blockIdx.x * 8;

    if (tid < 16) {
        const int tok = t0 + (tid >> 1), slot = tid & 1;
        sRowBase[tid] = (tok * EE + topk_idx[tok * 2 + slot]) * DD;
        sP[tid] = topk_p[tok * 2 + slot];
        sRowSq[tid] = 0.f;
    }
    if (tid < 32) sAttn[tid] = 0.f;
    __syncthreads();
    for (int i = tid; i < 16 * DD; i += 256) {
        const int r = i / DD, d = i - r * DD;
        sA[i] = f2bf(eo[(size_t)sRowBase[r] + d]);
    }
    __syncthreads();

    for (int h = 0; h < NHEAD; ++h) {
        for (int t = wave; t < 12; t += 8) {
            const int p = t >> 2, sub = t & 3;
            const int n0 = p * DD + h * HDIM + sub * 16;
            v8f acc = zero8();
            acc = gemm_tile(sA, DD, cinw, DD, n0, DD, 0, lane, acc);
            const int n = n0 + (lane & 15);
            const float bias = cinb[n];
            const int colb = p * HDIM + sub * 16 + (lane & 15);
            const int rowoff = (lane & 16) ? 8 : 0;
            for (int i = 0; i < 8; ++i)
                sQKV[(i + rowoff) * 192 + colb] = f2bf(acc[i] + bias);
        }
        __syncthreads();
        if (tid < 16) {                          // 8 tokens x 2 rows
            const int tk = tid >> 1, i = tid & 1;
            const bf16_t* q  = sQKV + (tk * 2 + i) * 192;
            const bf16_t* k0 = sQKV + (tk * 2 + 0) * 192 + 64;
            const bf16_t* k1 = sQKV + (tk * 2 + 1) * 192 + 64;
            float s0 = 0.f, s1 = 0.f;
            for (int c = 0; c < 64; ++c) {
                const float qc = bf2f(q[c]);
                s0 += qc * bf2f(k0[c]); s1 += qc * bf2f(k1[c]);
            }
            s0 *= 0.125f; s1 *= 0.125f;
            const float mx = fmaxf(s0, s1);
            const float e0 = __expf(s0 - mx), e1 = __expf(s1 - mx);
            const float inv = 1.f / (e0 + e1);
            const float p0 = e0 * inv, p1 = e1 * inv;
            sAttn[tid * 2 + 0] += p0 * (1.f / 12.f);     // head-mean attn
            sAttn[tid * 2 + 1] += p1 * (1.f / 12.f);
            bf16_t* orow = sO + (tk * 2 + i) * DD + h * HDIM;
            const bf16_t* v0 = sQKV + (tk * 2 + 0) * 192 + 128;
            const bf16_t* v1 = sQKV + (tk * 2 + 1) * 192 + 128;
            for (int c = 0; c < 64; ++c)
                orow[c] = f2bf(p0 * bf2f(v0[c]) + p1 * bf2f(v1[c]));
        }
        __syncthreads();
    }
    if (tid < 16) {                              // entropy of head-mean attention
        const float p0 = sAttn[tid * 2 + 0], p1 = sAttn[tid * 2 + 1];
        atomicAdd(entacc, -(p0 * __logf(p0 + 1e-8f) + p1 * __logf(p1 + 1e-8f)));
    }

    // out projection + residual + rmsnorm (collab overlays sA)
    bf16_t* sCol = sA;
    for (int t = wave; t < 48; t += 8) {
        const int n0 = t * 16;
        v8f acc = zero8();
        acc = gemm_tile(sO, DD, coutw, DD, n0, DD, 0, lane, acc);
        const int n = n0 + (lane & 15);
        const float bias = coutb[n];
        const int rowoff = (lane & 16) ? 8 : 0;
        for (int i = 0; i < 8; ++i) {
            const int r = i + rowoff;
            const float v = acc[i] + bias + eo[(size_t)sRowBase[r] + n];
            sCol[r * DD + n] = f2bf(v);
            atomicAdd(&sRowSq[r], v * v);
        }
    }
    __syncthreads();
    if (tid < 16) sScale[tid] = rsqrtf(sRowSq[tid] / (float)DD + 1e-5f);
    __syncthreads();
    for (int i = tid; i < 16 * DD; i += 256) {
        const int r = i / DD, d = i - r * DD;
        sCol[i] = f2bf(bf2f(sCol[i]) * sScale[r] * cnormw[d]);
    }
    __syncthreads();

    // refine MLP: r = relu(collab@f1^T+b)@f2^T + b + collab  (t1 chunked via LDS)
    bf16_t* sT1 = sO;
    v8f racc[6];
    for (int j = 0; j < 6; ++j) racc[j] = zero8();
    for (int hc = 0; hc < 2; ++hc) {
        const int hidb = hc * DD;
        for (int t = wave; t < 48; t += 8) {
            const int n0 = hidb + t * 16;
            v8f acc = zero8();
            acc = gemm_tile(sCol, DD, f1w, DD, n0, DD, 0, lane, acc);
            const int n = n0 + (lane & 15);
            const float bias = f1b[n];
            const int colb = t * 16 + (lane & 15);
            const int rowoff = (lane & 16) ? 8 : 0;
            for (int i = 0; i < 8; ++i) {
                float v = acc[i] + bias;
                v = v > 0.f ? v : 0.f;
                sT1[(i + rowoff) * DD + colb] = f2bf(v);
            }
        }
        __syncthreads();
        int j = 0;
        for (int t = wave; t < 48; t += 8, ++j)
            racc[j] = gemm_tile(sT1, DD, f2w, 2 * DD, t * 16, DD, hidb, lane, racc[j]);
        __syncthreads();
    }

    // combine with top-k probs -> y (rows 0-7 tokens, rows 8-15 zero), overlays sT1
    bf16_t* sY = sO;
    for (int i = tid; i < 8 * DD; i += 256) sY[8 * DD + i] = f2bf(0.f);
    {
        int j = 0;
        for (int t = wave; t < 48; t += 8, ++j) {
            const int n = t * 16 + (lane & 15);
            const int rowoff = (lane & 16) ? 8 : 0;
            const float bias = f2b[n];
            float rr[8];
            for (int i = 0; i < 8; ++i)
                rr[i] = racc[j][i] + bias + bf2f(sCol[(i + rowoff) * DD + n]);
            for (int tt = 0; tt < 4; ++tt) {
                const int r0 = rowoff + 2 * tt;
                const float y = sP[r0] * rr[2 * tt] + sP[r0 + 1] * rr[2 * tt + 1];
                const int ytok = (rowoff >> 1) + tt;       // 0-3 or 4-7
                sY[ytok * DD + n] = f2bf(y);
            }
        }
    }
    __syncthreads();

    // final = y @ o_w^T -> d_out
    for (int t = wave; t < 48; t += 8) {
        const int n0 = t * 16;
        v8f acc = zero8();
        acc = gemm_tile(sY, DD, oww, DD, n0, DD, 0, lane, acc);
        if (!(lane & 16)) {                                // rows 0-7 valid
            const int n = n0 + (lane & 15);
            for (int i = 0; i < 8; ++i)
                out[(size_t)(t0 + i) * DD + n] = acc[i];
        }
    }
}

// ---------------- finalize aux scalar ----------------
__global__ void finalize_kernel(const float* __restrict__ usage,
                                const float* __restrict__ ent,
                                float* __restrict__ outaux) {
    if (threadIdx.x == 0 && blockIdx.x == 0) {
        float u[8], mean = 0.f;
        for (int e = 0; e < 8; ++e) { u[e] = usage[e] / (float)BS; mean += u[e]; }
        mean *= 0.125f;
        float var = 0.f;
        for (int e = 0; e < 8; ++e) { const float d = u[e] - mean; var += d * d; }
        var /= 7.f;                                        // ddof=1
        const float e_ = ent[0] / (float)(BS * 2);
        outaux[0] = (0.01f * (-e_) + 0.01f * var) * 0.01f; // AUX_W = 0.01
    }
}

// ---------------- host orchestration ----------------
static void cast_launch(const float* src, bf16_t* dst, long n, hipStream_t s) {
    long blocks = (n + 255) / 256;
    if (blocks > 4096) blocks = 4096;
    cast_f32_bf16<<<(int)blocks, 256, 0, s>>>(src, dst, n);
}

extern "C" void kernel_launch(void* const* d_in, const int* in_sizes, int n_in,
                              void* d_out, int out_size, void* d_ws, size_t ws_size,
                              hipStream_t stream) {
    (void)in_sizes; (void)n_in; (void)out_size; (void)ws_size;
    const float* x      = (const float*)d_in[0];
    const float* w1     = (const float*)d_in[1];
    const float* w3     = (const float*)d_in[2];
    const float* w2     = (const float*)d_in[3];
    const float* r_in_w = (const float*)d_in[4];
    const float* r_in_b = (const float*)d_in[5];
    const float* r_out_w= (const float*)d_in[6];
    const float* r_out_b= (const float*)d_in[7];
    const float* r_norm = (const float*)d_in[8];
    const float* gate_w = (const float*)d_in[9];
    const float* gate_b = (const float*)d_in[10];
    const float* c_in_w = (const float*)d_in[11];
    const float* c_in_b = (const float*)d_in[12];
    const float* c_out_w= (const float*)d_in[13];
    const float* c_out_b= (const float*)d_in[14];
    const float* c_norm = (const float*)d_in[15];
    const float* f1_w   = (const float*)d_in[16];
    const float* f1_b   = (const float*)d_in[17];
    const float* f2_w   = (const float*)d_in[18];
    const float* f2_b   = (const float*)d_in[19];
    const float* o_w    = (const float*)d_in[20];

    char* ws = (char*)d_ws;
    size_t off = 0;
    auto take = [&](size_t bytes) -> char* {
        char* p = ws + off;
        off = (off + bytes + 255) & ~(size_t)255;
        return p;
    };
    bf16_t* xbf    = (bf16_t*)take((size_t)BS * DD * 2);
    bf16_t* w1bf   = (bf16_t*)take((size_t)EE * HH * DD * 2);
    bf16_t* w3bf   = (bf16_t*)take((size_t)EE * HH * DD * 2);
    bf16_t* w2bf   = (bf16_t*)take((size_t)EE * DD * HH * 2);
    bf16_t* rinwbf = (bf16_t*)take((size_t)TD3 * DD * 2);
    bf16_t* routwbf= (bf16_t*)take((size_t)DD * DD * 2);
    bf16_t* cinwbf = (bf16_t*)take((size_t)TD3 * DD * 2);
    bf16_t* coutwbf= (bf16_t*)take((size_t)DD * DD * 2);
    bf16_t* f1bf   = (bf16_t*)take((size_t)2 * DD * DD * 2);
    bf16_t* f2bf_w = (bf16_t*)take((size_t)DD * 2 * DD * 2);
    bf16_t* owbf   = (bf16_t*)take((size_t)DD * DD * 2);
    float*  eo     = (float*)take((size_t)BS * EE * DD * 4);
    int*    tki    = (int*)take((size_t)BS * 2 * 4);
    float*  tkp    = (float*)take((size_t)BS * 2 * 4);
    float*  stats  = (float*)take(16 * 4);      // usage[8], ent[1]

    cast_launch(x,      xbf,    (long)BS * DD,      stream);
    cast_launch(w1,     w1bf,   (long)EE * HH * DD, stream);
    cast_launch(w3,     w3bf,   (long)EE * HH * DD, stream);
    cast_launch(w2,     w2bf,   (long)EE * DD * HH, stream);
    cast_launch(r_in_w, rinwbf, (long)TD3 * DD,     stream);
    cast_launch(r_out_w,routwbf,(long)DD * DD,      stream);
    cast_launch(c_in_w, cinwbf, (long)TD3 * DD,     stream);
    cast_launch(c_out_w,coutwbf,(long)DD * DD,      stream);
    cast_launch(f1_w,   f1bf,   (long)2 * DD * DD,  stream);
    cast_launch(f2_w,   f2bf_w, (long)DD * 2 * DD,  stream);
    cast_launch(o_w,    owbf,   (long)DD * DD,      stream);
    zero_stats_kernel<<<1, 32, 0, stream>>>(stats);

    ffn_kernel<<<dim3(BS / 32, EE), 256, 0, stream>>>(xbf, w1bf, w3bf, w2bf, eo);

    router_kernel<<<BS / 2, 256, 0, stream>>>(eo, rinwbf, r_in_b, routwbf, r_out_b,
                                              r_norm, gate_w, gate_b, tki, tkp, stats);

    collab_kernel<<<BS / 8, 256, 0, stream>>>(eo, tki, tkp, cinwbf, c_in_b, coutwbf,
                                              c_out_b, c_norm, f1bf, f1_b, f2bf_w,
                                              f2_b, owbf, (float*)d_out, stats + 8);

    finalize_kernel<<<1, 1, 0, stream>>>(stats, stats + 8,
                                         (float*)d_out + (size_t)BS * DD);
}
